// MultiheadAttention_31834297598614
// MI455X (gfx1250) — matrix operand order repaired and checked by reading, not compiled, not run on hardware
//
#include <hip/hip_runtime.h>
#include <hip/hip_bf16.h>
#include <stdint.h>

// Shapes from the reference
#define Bb 8
#define Ss 128
#define Ee 768
#define Hh 12
#define Dd 64
#define SD 8192            // S*D
#define MROT 192           // 2*B*H rows (q rows 0..95, k rows 96..191)

typedef __attribute__((ext_vector_type(16))) __bf16 v16bf;
typedef __attribute__((ext_vector_type(8)))  __bf16 v8bf;
typedef __attribute__((ext_vector_type(8)))  float  v8f;

static __device__ __forceinline__ v16bf cat16(v8bf lo, v8bf hi) {
  return __builtin_shufflevector(lo, hi, 0,1,2,3,4,5,6,7,8,9,10,11,12,13,14,15);
}

static __device__ __forceinline__ v8f wmma_bf16(v16bf a, v16bf b, v8f c) {
  return __builtin_amdgcn_wmma_f32_16x16x32_bf16(false, a, false, b, (short)0, c, false, false);
}

// CDNA5 async global->LDS copy (pure copy path, ASYNCcnt-tracked).
// VDST = LDS byte address (VGPR), VADDR = 64-bit global address.
static __device__ __forceinline__ void async_ld_b128(uint32_t lds_off, const void* gaddr) {
  asm volatile("global_load_async_to_lds_b128 %0, %1, off"
               :: "v"(lds_off), "v"(gaddr) : "memory");
}
static __device__ __forceinline__ void wait_async0() {
  asm volatile("s_wait_asynccnt 0" ::: "memory");
}
static __device__ __forceinline__ uint32_t lds_addr(const void* p) {
  return (uint32_t)(uintptr_t)p;   // low 32 bits of generic ptr = LDS byte offset
}

// ---------------------------------------------------------------------------
// Kernel: f32 -> bf16 elementwise (hidden_states)
// ---------------------------------------------------------------------------
__global__ void cvt_bf16_kernel(const float* __restrict__ in, __bf16* __restrict__ out, int n) {
  int i = blockIdx.x * 256 + threadIdx.x;
  if (i < n) out[i] = (__bf16)in[i];
}

// ---------------------------------------------------------------------------
// Kernel: transpose + convert the four 768x768 weights:  Wt[z][e][o] = W_z[o][e]
// grid (24,24,4), block (32,8)
// ---------------------------------------------------------------------------
__global__ void wt_kernel(const float* __restrict__ Wq, const float* __restrict__ Wk,
                          const float* __restrict__ Wv, const float* __restrict__ Wo,
                          __bf16* __restrict__ Wt) {
  __shared__ float tile[32][33];
  const int z = blockIdx.z;
  const float* W = (z == 0) ? Wq : (z == 1) ? Wk : (z == 2) ? Wv : Wo;
  const int bx = blockIdx.x * 32;   // e base
  const int by = blockIdx.y * 32;   // o base
  const int tx = threadIdx.x, ty = threadIdx.y;
  #pragma unroll
  for (int j = 0; j < 32; j += 8)
    tile[ty + j][tx] = W[(size_t)(by + ty + j) * Ee + (bx + tx)];
  __syncthreads();
  __bf16* dst = Wt + (size_t)z * Ee * Ee;
  #pragma unroll
  for (int j = 0; j < 32; j += 8)
    dst[(size_t)(bx + ty + j) * Ee + (by + tx)] = (__bf16)tile[tx][ty + j];
}

// ---------------------------------------------------------------------------
// Projection GEMM: [1024,768] @ Wt_z[768,768] (z=0 q, 1 k, 2 v), WMMA bf16.
// B (bf16 weights) staged to LDS with async global->LDS b128, double buffered.
// q/k scattered (bf16) into X[192][8192] rotation layout; v stored f32 [B,S,H,D].
// grid (12, 8, 3)  block 256 (8 waves, each a 16x64 tile; M slab 128, N slab 64)
// ---------------------------------------------------------------------------
__global__ __launch_bounds__(256, 1) void proj_kernel(
    const __bf16* __restrict__ A,   // hs bf16 [1024][768]
    const __bf16* __restrict__ Wt,  // [3][768][768]
    __bf16* __restrict__ X,         // [192][8192]
    float* __restrict__ Vbuf) {     // [1024][768]
  __shared__ __bf16 lds[2][32][72];
  const int tid = threadIdx.x;
  const int wave = tid >> 5, lane = tid & 31;
  const int lane16 = lane & 15, halfSel = lane >> 4;
  const int z = blockIdx.z;
  const int n0 = blockIdx.x * 64;
  const int m0 = blockIdx.y * 128 + wave * 16;
  const __bf16* Bm = Wt + (size_t)z * Ee * Ee;

  // one async b128 per thread stages a full 32x64 bf16 block (4 KB)
  const int sr = tid >> 3;            // 0..31  (K row)
  const int sc = (tid & 7) * 8;       // bf16 col, 8 elems = 16 B
  auto stage = [&](int kk, int buf) {
    async_ld_b128(lds_addr(&lds[buf][sr][sc]),
                  Bm + (size_t)(kk + sr) * Ee + n0 + sc);
  };

  v8f acc[4] = {};
  const int mrow = m0 + lane16;
  stage(0, 0);
  wait_async0();
  __syncthreads();
  for (int kk = 0; kk < Ee; kk += 32) {
    const int buf = (kk >> 5) & 1;
    if (kk + 32 < Ee) stage(kk + 32, buf ^ 1);

    const __bf16* ap = A + (size_t)mrow * Ee + kk + 8 * halfSel;
    v16bf afrag = cat16(*(const v8bf*)ap, *(const v8bf*)(ap + 16));
    #pragma unroll
    for (int t = 0; t < 4; ++t) {
      v16bf bfrag;
      #pragma unroll
      for (int s = 0; s < 16; ++s)
        bfrag[s] = lds[buf][(s & 7) + 8 * halfSel + 16 * (s >> 3)][t * 16 + lane16];
      acc[t] = wmma_bf16(afrag, bfrag, acc[t]);
    }
    wait_async0();
    __syncthreads();
  }
  #pragma unroll
  for (int t = 0; t < 4; ++t) {
    #pragma unroll
    for (int r = 0; r < 8; ++r) {
      const int i = m0 + halfSel * 8 + r;          // 0..1023  (b*128+s)
      const int o = n0 + t * 16 + lane16;          // 0..767   (h*64+d)
      const float val = acc[t][r];
      if (z == 2) {
        Vbuf[(size_t)i * Ee + o] = val;
      } else {
        const int row = (z ? 96 : 0) + (i >> 7) * Hh + (o >> 6);  // b*12+h
        const int col = (i & 127) * Dd + (o & 63);                // s*64+d
        X[(size_t)row * SD + col] = (__bf16)val;
      }
    }
  }
}

// ---------------------------------------------------------------------------
// Rotation GEMM: Cpart[kb][192,8192] = X[192, khalf] @ R[khalf, 8192]
// 6 waves/WG; each wave = 2 M-tiles x 4 N-tiles (32x64). WG covers M=192,
// N slab 64, K half 4096. R staged f32->bf16 once per K-step (conversion must
// pass through VGPRs, so plain loads beat async here), double-buffered LDS.
// R is streamed from HBM exactly once. grid (128, 2), block 192
// ---------------------------------------------------------------------------
__global__ __launch_bounds__(192, 1) void rot_gemm_kernel(
    const float* __restrict__ R, const __bf16* __restrict__ X,
    float* __restrict__ Cpart) {
  __shared__ __bf16 lds[2][32][72];   // 32 K-rows x 64 N-cols, padded to 72
  const int tid = threadIdx.x;
  const int wave = tid >> 5, lane = tid & 31;
  const int lane16 = lane & 15, halfSel = lane >> 4;
  const int n0 = blockIdx.x * 64;
  const int k0 = blockIdx.y * (SD / 2);
  float* C = Cpart + (size_t)blockIdx.y * MROT * SD;

  auto stage = [&](int kk, int buf) {
    #pragma unroll
    for (int it = 0; it < 3; ++it) {
      const int idx = tid + it * 192;
      if (idx < 512) {                       // 32 rows x 16 float4 per row
        const int r = idx >> 4;
        const int c = (idx & 15) << 2;
        const float4 f = *reinterpret_cast<const float4*>(
            R + (size_t)(kk + r) * SD + (n0 + c));
        lds[buf][r][c + 0] = (__bf16)f.x;
        lds[buf][r][c + 1] = (__bf16)f.y;
        lds[buf][r][c + 2] = (__bf16)f.z;
        lds[buf][r][c + 3] = (__bf16)f.w;
      }
    }
  };

  v8f acc[2][4] = {};
  stage(k0, 0);
  __syncthreads();
  for (int kk = k0; kk < k0 + SD / 2; kk += 32) {
    const int buf = (kk >> 5) & 1;
    if (kk + 32 < k0 + SD / 2) stage(kk + 32, buf ^ 1);

    v16bf afrag[2];
    #pragma unroll
    for (int u = 0; u < 2; ++u) {
      const __bf16* ap =
          X + (size_t)(wave * 32 + u * 16 + lane16) * SD + kk + 8 * halfSel;
      afrag[u] = cat16(*(const v8bf*)ap, *(const v8bf*)(ap + 16));
    }
    #pragma unroll
    for (int t = 0; t < 4; ++t) {
      v16bf bfrag;
      #pragma unroll
      for (int s = 0; s < 16; ++s)
        bfrag[s] = lds[buf][(s & 7) + 8 * halfSel + 16 * (s >> 3)][t * 16 + lane16];
      acc[0][t] = wmma_bf16(afrag[0], bfrag, acc[0][t]);
      acc[1][t] = wmma_bf16(afrag[1], bfrag, acc[1][t]);
    }
    __syncthreads();
  }
  #pragma unroll
  for (int u = 0; u < 2; ++u) {
    #pragma unroll
    for (int t = 0; t < 4; ++t) {
      #pragma unroll
      for (int r = 0; r < 8; ++r) {
        const int row = wave * 32 + u * 16 + halfSel * 8 + r;
        const int col = n0 + t * 16 + lane16;
        C[(size_t)row * SD + col] = acc[u][t][r];
      }
    }
  }
}

// ---------------------------------------------------------------------------
// Attention over the HEAD axis (12x12), softmax, @ V. One block per (b,s).
// Sums the two K-split partials of q_rot/k_rot on load. grid 1024, block 192
// ---------------------------------------------------------------------------
__global__ __launch_bounds__(192, 1) void attn_kernel(
    const float* __restrict__ P,      // Cpart: [2][192][8192]
    const float* __restrict__ Vbuf,   // [1024][768]
    const float* __restrict__ mask,   // [8][1][12][12]
    float* __restrict__ attn_out,     // [1024][12][12]
    __bf16* __restrict__ mid) {       // [B][H][S][D] flat == reshaped [1024][768]
  const int bs = blockIdx.x;
  const int b = bs >> 7, s = bs & 127;
  __shared__ float q[Hh][Dd], k[Hh][Dd], v[Hh][Dd], sc[Hh][Hh];
  const int tid = threadIdx.x;
  for (int idx = tid; idx < Hh * Dd; idx += 192) {
    const int h = idx >> 6, d = idx & 63;
    const size_t qo = (size_t)(b * Hh + h) * SD + s * Dd + d;
    const size_t ko = (size_t)(96 + b * Hh + h) * SD + s * Dd + d;
    q[h][d] = P[qo] + P[(size_t)MROT * SD + qo];
    k[h][d] = P[ko] + P[(size_t)MROT * SD + ko];
    v[h][d] = Vbuf[(size_t)bs * Ee + idx];
  }
  __syncthreads();
  if (tid < Hh * Hh) {
    const int h = tid / Hh, g = tid % Hh;
    float a = 0.f;
    #pragma unroll 8
    for (int d = 0; d < Dd; ++d) a += q[h][d] * k[g][d];
    sc[h][g] = a * 0.125f + mask[(size_t)b * (Hh * Hh) + h * Hh + g];
  }
  __syncthreads();
  if (tid < Hh) {
    const int h = tid;
    float m = sc[h][0];
    for (int g = 1; g < Hh; ++g) m = fmaxf(m, sc[h][g]);
    float e[Hh], sum = 0.f;
    for (int g = 0; g < Hh; ++g) { e[g] = __expf(sc[h][g] - m); sum += e[g]; }
    const float inv = 1.0f / sum;
    for (int g = 0; g < Hh; ++g) {
      const float p = e[g] * inv;
      sc[h][g] = p;
      attn_out[(size_t)bs * (Hh * Hh) + h * Hh + g] = p;
    }
  }
  __syncthreads();
  for (int idx = tid; idx < Hh * Dd; idx += 192) {
    const int h = idx >> 6, d = idx & 63;
    float a = 0.f;
    #pragma unroll
    for (int g = 0; g < Hh; ++g) a += sc[h][g] * v[g][d];
    mid[(size_t)b * (Hh * SD) + (size_t)h * SD + s * Dd + d] = (__bf16)a;
  }
}

// ---------------------------------------------------------------------------
// Final GEMM: out[1024,768] = mid[1024,768] @ WoT[768,768], f32 output.
// Same async-staged structure as proj_kernel. grid (12, 8), block 256
// ---------------------------------------------------------------------------
__global__ __launch_bounds__(256, 1) void final_gemm_kernel(
    const __bf16* __restrict__ A, const __bf16* __restrict__ Bm,
    float* __restrict__ out) {
  __shared__ __bf16 lds[2][32][72];
  const int tid = threadIdx.x;
  const int wave = tid >> 5, lane = tid & 31;
  const int lane16 = lane & 15, halfSel = lane >> 4;
  const int n0 = blockIdx.x * 64;
  const int m0 = blockIdx.y * 128 + wave * 16;

  const int sr = tid >> 3;
  const int sc = (tid & 7) * 8;
  auto stage = [&](int kk, int buf) {
    async_ld_b128(lds_addr(&lds[buf][sr][sc]),
                  Bm + (size_t)(kk + sr) * Ee + n0 + sc);
  };

  v8f acc[4] = {};
  const int mrow = m0 + lane16;
  stage(0, 0);
  wait_async0();
  __syncthreads();
  for (int kk = 0; kk < Ee; kk += 32) {
    const int buf = (kk >> 5) & 1;
    if (kk + 32 < Ee) stage(kk + 32, buf ^ 1);

    const __bf16* ap = A + (size_t)mrow * Ee + kk + 8 * halfSel;
    v16bf afrag = cat16(*(const v8bf*)ap, *(const v8bf*)(ap + 16));
    #pragma unroll
    for (int t = 0; t < 4; ++t) {
      v16bf bfrag;
      #pragma unroll
      for (int s = 0; s < 16; ++s)
        bfrag[s] = lds[buf][(s & 7) + 8 * halfSel + 16 * (s >> 3)][t * 16 + lane16];
      acc[t] = wmma_bf16(afrag, bfrag, acc[t]);
    }
    wait_async0();
    __syncthreads();
  }
  #pragma unroll
  for (int t = 0; t < 4; ++t) {
    #pragma unroll
    for (int r = 0; r < 8; ++r) {
      const int i = m0 + halfSel * 8 + r;
      const int o = n0 + t * 16 + lane16;
      out[(size_t)i * Ee + o] = acc[t][r];
    }
  }
}

// ---------------------------------------------------------------------------
extern "C" void kernel_launch(void* const* d_in, const int* in_sizes, int n_in,
                              void* d_out, int out_size, void* d_ws, size_t ws_size,
                              hipStream_t stream) {
  const float* hs   = (const float*)d_in[0];
  const float* R    = (const float*)d_in[1];
  const float* mask = (const float*)d_in[2];
  const float* Wq   = (const float*)d_in[3];
  const float* Wk   = (const float*)d_in[4];
  const float* Wv   = (const float*)d_in[5];
  const float* Wo   = (const float*)d_in[6];
  float* out = (float*)d_out;

  char* ws = (char*)d_ws;
  __bf16* hs_bf = (__bf16*)(ws + 0);          // 1024*768*2          = 1.5 MB
  __bf16* Wt    = (__bf16*)(ws + 1572864);    // 4*768*768*2         = 4.5 MB
  __bf16* X     = (__bf16*)(ws + 6291456);    // 192*8192*2          = 3.0 MB
  float*  Vbuf  = (float*) (ws + 9437184);    // 1024*768*4          = 3.0 MB
  float*  Cpart = (float*) (ws + 12582912);   // 2*192*8192*4        = 12.0 MB
  __bf16* mid   = (__bf16*)(ws + 25165824);   // 1024*768*2          = 1.5 MB

  cvt_bf16_kernel<<<3072, 256, 0, stream>>>(hs, hs_bf, Bb * Ss * Ee);
  wt_kernel<<<dim3(24, 24, 4), dim3(32, 8), 0, stream>>>(Wq, Wk, Wv, Wo, Wt);
  proj_kernel<<<dim3(12, 8, 3), 256, 0, stream>>>(hs_bf, Wt, X, Vbuf);
  rot_gemm_kernel<<<dim3(128, 2), 192, 0, stream>>>(R, X, Cpart);
  attn_kernel<<<1024, 192, 0, stream>>>(Cpart, Vbuf, mask,
                                        out + (size_t)Bb * Ss * Ee, mid);
  final_gemm_kernel<<<dim3(12, 8), 256, 0, stream>>>(
      mid, Wt + (size_t)3 * Ee * Ee, out);
}
